// SupervisedGroundedCoreferencer_463856468557
// MI455X (gfx1250) — matrix operand order repaired and checked by reading, not compiled, not run on hardware
//
#include <hip/hip_runtime.h>

// ---------- types ----------
typedef __attribute__((ext_vector_type(16))) __bf16 bf16x16;
typedef __attribute__((ext_vector_type(8)))  float  f32x8;

union FragB { uint4 q[2]; bf16x16 v; };
union AccU  { f32x8 v; float f[8]; };

__device__ __forceinline__ unsigned short f2bf(float x) {
    unsigned int u = __float_as_uint(x);
    u += 0x7FFFu + ((u >> 16) & 1u);          // round-to-nearest-even
    return (unsigned short)(u >> 16);
}

// ---------- pack / convert kernels ----------
__global__ void k_f32_to_bf16(const float* __restrict__ src,
                              unsigned short* __restrict__ dst, int n) {
    int i = blockIdx.x * blockDim.x + threadIdx.x;
    if (i < n) dst[i] = f2bf(src[i]);
}

// Pack a K x 1024 (row-major f32) weight matrix into WMMA-B lane layout, bf16.
// Block (nt, kt) covers K rows [kt*32, kt*32+32), cols [nt*16, nt*16+16).
// Lane l holds column n = l&15, K values kb2..kb2+15 with kb2 = (l>>4)*16.
// ktShift = log2(K/32)  (K is 512 -> 4, 1024 -> 5).
__global__ void k_pack_b(const float* __restrict__ src,
                         unsigned short* __restrict__ dst, int ktShift) {
    int idx = blockIdx.x * blockDim.x + threadIdx.x;
    int total = (32 << ktShift) * 1024;
    if (idx >= total) return;
    int c   = idx & 15;
    int l   = (idx >> 4) & 31;
    int blk = idx >> 9;
    int kt  = blk & ((1 << ktShift) - 1);
    int nt  = blk >> ktShift;
    int k   = kt * 32 + ((l >> 4) * 16) + c;
    int n   = nt * 16 + (l & 15);
    dst[idx] = f2bf(src[k * 1024 + n]);
}

// ---------- hf / hs GEMM: (256x512 bf16) @ packed(512x1024) -> f32 ----------
// 2048 tiles total: which(2) x mt(16) x nt(64). One wave per 16x16 tile.
__global__ __launch_bounds__(256) void k_hfhs(
    const unsigned short* __restrict__ fB, const unsigned short* __restrict__ gB,
    const unsigned short* __restrict__ W1fP, const unsigned short* __restrict__ W1sP,
    float* __restrict__ hf, float* __restrict__ hs) {
    int wave = (blockIdx.x * blockDim.x + threadIdx.x) >> 5;
    int lane = threadIdx.x & 31;
    int which = wave >> 10;          // 0 -> hf, 1 -> hs
    int t  = wave & 1023;
    int mt = t >> 6;                 // 0..15
    int nt = t & 63;                 // 0..63
    const unsigned short* A  = which ? gB   : fB;
    const unsigned short* Bp = which ? W1sP : W1fP;
    float* Cout = which ? hs : hf;

    int mA  = lane & 15;
    int kbA = (lane >> 4) * 16;      // contiguous k = 16h + s, same order as packed B
    const uint4* ap = (const uint4*)(A + (mt * 16 + mA) * 512 + kbA);
    const uint4* bp = (const uint4*)(Bp + nt * 8192 + lane * 16);

    AccU acc; acc.v = {};
    #pragma unroll
    for (int kt = 0; kt < 16; ++kt) {
        FragB a, b;
        a.q[0] = ap[kt * 4];
        a.q[1] = ap[kt * 4 + 1];
        b.q[0] = bp[kt * 64];
        b.q[1] = bp[kt * 64 + 1];
        acc.v = __builtin_amdgcn_wmma_f32_16x16x32_bf16(
            false, a.v, false, b.v, (short)0, acc.v, false, false);
    }
    int col = nt * 16 + (lane & 15);
    int rb  = (lane >> 4) * 8;
    #pragma unroll
    for (int r = 0; r < 8; ++r) {
        int row = mt * 16 + rb + r;
        Cout[row * 1024 + col] = acc.f[r];
    }
}

// ---------- fused main kernel ----------
// One block per (i, j0..j0+31): builds P = f_i * g_j (bf16, LDS),
// h = relu(P@W1p + hf_i + hs_j + b1) kept in LDS (bf16),
// scores = relu(h@W2 + b2) @ W3 + b3, fused via per-column accumulation.
// Register-blocked: each wave keeps 8 accumulator tiles live, reuses one A
// fragment across 8 WMMAs, and all fragment addresses fold into immediate
// offsets off hoisted base pointers.
__global__ __launch_bounds__(256) void k_main(
    const float* __restrict__ fF, const float* __restrict__ gF,
    const float* __restrict__ hf, const float* __restrict__ hs,
    const float* __restrict__ b1, const float* __restrict__ b2,
    const float* __restrict__ W3, const float* __restrict__ b3,
    const float* __restrict__ span_mask, const float* __restrict__ image_mask,
    const unsigned short* __restrict__ W1pP, const unsigned short* __restrict__ W2P,
    float* __restrict__ scores) {
    __shared__ float fi[512];                  //  2 KB
    __shared__ float hfL[1024];                //  4 KB
    __shared__ float b1L[1024];                //  4 KB
    __shared__ unsigned short P[32 * 512];     // 32 KB
    __shared__ unsigned short Hh[32 * 1024];   // 64 KB
    __shared__ float scoreL[32];

    int tid = threadIdx.x;
    int p0  = blockIdx.x * 32;                  // pair base
    int i   = p0 >> 8;                          // span row
    int j0  = p0 & 255;                         // image row base

    // phase 0: stage fi, hf_i, b1; zero score accumulators
    for (int t = tid; t < 512; t += 256) fi[t] = fF[i * 512 + t];
    for (int t = tid; t < 1024; t += 256) {
        hfL[t] = hf[i * 1024 + t];
        b1L[t] = b1[t];
    }
    if (tid < 32) scoreL[tid] = 0.0f;
    __syncthreads();

    // phase 1: P[row][k] = bf16(fi[k] * g[j0+row][k]); 32 rows x 512
    {
        int row = tid >> 3;
        int k0  = (tid & 7) * 64;
        const float4* gp = (const float4*)(gF + (j0 + row) * 512 + k0);
        unsigned short* dst = &P[row * 512 + k0];
        #pragma unroll
        for (int q = 0; q < 16; ++q) {
            float4 gv = gp[q];
            int k = k0 + q * 4;
            uint2 pk;
            pk.x = (unsigned)f2bf(gv.x * fi[k + 0]) |
                   ((unsigned)f2bf(gv.y * fi[k + 1]) << 16);
            pk.y = (unsigned)f2bf(gv.z * fi[k + 2]) |
                   ((unsigned)f2bf(gv.w * fi[k + 3]) << 16);
            *(uint2*)(dst + q * 4) = pk;
        }
    }
    __syncthreads();

    int wv    = tid >> 5;
    int lane  = tid & 31;
    int mA    = lane & 15;
    int kbA   = (lane >> 4) * 16;
    int col16 = lane & 15;
    int rb    = (lane >> 4) * 8;
    int m_t   = wv & 1;                 // which 16-row half
    int ntb   = (wv >> 1) * 16;         // 16 n-tiles per wave, 2 groups of 8

    // phase 2: GEMM1  h = relu(P @ W1p + hf + hs + b1) -> Hh (bf16, LDS)
    {
        const uint4* ap = (const uint4*)(&P[(m_t * 16 + mA) * 512] + kbA);
        for (int g = 0; g < 2; ++g) {
            // 8 consecutive n-tile blocks; block stride 512 elems (64 uint4),
            // n-tile stride 16 blocks (1024 uint4).
            const uint4* bp =
                (const uint4*)(W1pP + (ntb + g * 8) * 8192 + lane * 16);
            AccU acc[8];
            #pragma unroll
            for (int u = 0; u < 8; ++u) acc[u].v = {};
            #pragma unroll 4
            for (int kt = 0; kt < 16; ++kt) {
                FragB a;
                a.q[0] = ap[kt * 4];
                a.q[1] = ap[kt * 4 + 1];
                #pragma unroll
                for (int u = 0; u < 8; ++u) {
                    FragB b;
                    b.q[0] = bp[u * 1024 + kt * 64];
                    b.q[1] = bp[u * 1024 + kt * 64 + 1];
                    acc[u].v = __builtin_amdgcn_wmma_f32_16x16x32_bf16(
                        false, a.v, false, b.v, (short)0, acc[u].v, false, false);
                }
            }
            #pragma unroll
            for (int u = 0; u < 8; ++u) {
                int col = (ntb + g * 8 + u) * 16 + col16;
                float hfv = hfL[col];
                float b1v = b1L[col];
                #pragma unroll
                for (int r = 0; r < 8; ++r) {
                    int rowL = m_t * 16 + rb + r;
                    float v = acc[u].f[r] + hfv + hs[(j0 + rowL) * 1024 + col] + b1v;
                    Hh[rowL * 1024 + col] = f2bf(fmaxf(v, 0.0f));
                }
            }
        }
    }
    __syncthreads();

    // phase 3: GEMM2 fused with W3 reduction
    {
        const uint4* ap = (const uint4*)(&Hh[(m_t * 16 + mA) * 1024] + kbA);
        float sc[8] = {0.f, 0.f, 0.f, 0.f, 0.f, 0.f, 0.f, 0.f};
        for (int g = 0; g < 2; ++g) {
            // block stride 512 elems (64 uint4), n-tile stride 32 blocks
            // (2048 uint4).
            const uint4* bp =
                (const uint4*)(W2P + (ntb + g * 8) * 16384 + lane * 16);
            AccU acc[8];
            #pragma unroll
            for (int u = 0; u < 8; ++u) acc[u].v = {};
            #pragma unroll 4
            for (int kt = 0; kt < 32; ++kt) {
                FragB a;
                a.q[0] = ap[kt * 4];
                a.q[1] = ap[kt * 4 + 1];
                #pragma unroll
                for (int u = 0; u < 8; ++u) {
                    FragB b;
                    b.q[0] = bp[u * 2048 + kt * 64];
                    b.q[1] = bp[u * 2048 + kt * 64 + 1];
                    acc[u].v = __builtin_amdgcn_wmma_f32_16x16x32_bf16(
                        false, a.v, false, b.v, (short)0, acc[u].v, false, false);
                }
            }
            #pragma unroll
            for (int u = 0; u < 8; ++u) {
                int col = (ntb + g * 8 + u) * 16 + col16;
                float b2v = b2[col];
                float w3v = W3[col];
                #pragma unroll
                for (int r = 0; r < 8; ++r) {
                    float h2 = fmaxf(acc[u].f[r] + b2v, 0.0f);
                    sc[r] += h2 * w3v;
                }
            }
        }
        #pragma unroll
        for (int r = 0; r < 8; ++r)
            atomicAdd(&scoreL[m_t * 16 + rb + r], sc[r]);
    }
    __syncthreads();

    // phase 4: scores out (+b3, *mask)
    if (tid < 32) {
        int p = p0 + tid;
        float val = scoreL[tid] + b3[0];
        float m = span_mask[p >> 8] * image_mask[p & 255];
        scores[p] = val * m;
    }
}

// ---------- S / loss reduction (single block, 256 threads) ----------
__global__ __launch_bounds__(256) void k_loss(const float* __restrict__ scores,
                                              float* __restrict__ out) {
    __shared__ float Sv[256];
    __shared__ float lseR[16];
    __shared__ float lseC[16];
    int t = threadIdx.x;
    {   // S[i] = max over 16 groups of (sum of 16 consecutive scores) in row i
        const float* row = scores + t * 256;
        float best = -3.4e38f;
        for (int g = 0; g < 16; ++g) {
            float s = 0.0f;
            #pragma unroll
            for (int d = 0; d < 16; ++d) s += row[g * 16 + d];
            best = fmaxf(best, s);
        }
        Sv[t] = best;
    }
    __syncthreads();
    if (t < 16) {                       // row logsumexp of the 16x16 S
        float m = -3.4e38f;
        for (int c = 0; c < 16; ++c) m = fmaxf(m, Sv[t * 16 + c]);
        float s = 0.0f;
        for (int c = 0; c < 16; ++c) s += expf(Sv[t * 16 + c] - m);
        lseR[t] = m + logf(s);
    } else if (t < 32) {                // column logsumexp
        int c = t - 16;
        float m = -3.4e38f;
        for (int r = 0; r < 16; ++r) m = fmaxf(m, Sv[r * 16 + c]);
        float s = 0.0f;
        for (int r = 0; r < 16; ++r) s += expf(Sv[r * 16 + c] - m);
        lseC[c] = m + logf(s);
    }
    __syncthreads();
    if (t == 0) {
        float sumS = 0.0f;
        for (int k = 0; k < 256; ++k) sumS += Sv[k];
        float sR = 0.0f, sC = 0.0f;
        for (int k = 0; k < 16; ++k) { sR += lseR[k]; sC += lseC[k]; }
        float t1 = sumS - 16.0f * sR;   // sum(log_softmax(S, axis=1))
        float t2 = sumS - 16.0f * sC;   // sum(log_softmax(S.T, axis=1))
        out[0] = -(t1 + t2) / 16.0f;
    }
}

// ---------- host launch ----------
extern "C" void kernel_launch(void* const* d_in, const int* in_sizes, int n_in,
                              void* d_out, int out_size, void* d_ws, size_t ws_size,
                              hipStream_t stream) {
    (void)in_sizes; (void)n_in; (void)out_size; (void)ws_size;
    const float* spanE  = (const float*)d_in[0];   // (16,16,512)
    const float* imageE = (const float*)d_in[1];   // (16,16,512)
    const float* spanM  = (const float*)d_in[2];   // (16,16)
    const float* imageM = (const float*)d_in[3];   // (16,16)
    const float* W1     = (const float*)d_in[4];   // (1536,1024)
    const float* b1     = (const float*)d_in[5];   // (1024,)
    const float* W2     = (const float*)d_in[6];   // (1024,1024)
    const float* b2     = (const float*)d_in[7];   // (1024,)
    const float* W3     = (const float*)d_in[8];   // (1024,1)
    const float* b3     = (const float*)d_in[9];   // (1,)

    // workspace layout
    unsigned short* fB   = (unsigned short*)d_ws;          // 256*512
    unsigned short* gB   = fB   + 256 * 512;               // 256*512
    unsigned short* W1fP = gB   + 256 * 512;               // 512*1024
    unsigned short* W1sP = W1fP + 512 * 1024;
    unsigned short* W1pP = W1sP + 512 * 1024;
    unsigned short* W2P  = W1pP + 512 * 1024;              // 1024*1024
    float* hf = (float*)(W2P + 1024 * 1024);               // 256*1024 f32
    float* hs = hf + 256 * 1024;                           // 256*1024 f32

    float* loss_out   = (float*)d_out;
    float* scores_out = loss_out + 1;                      // 65536 f32

    // 1. convert activations to bf16
    k_f32_to_bf16<<<(256 * 512 + 255) / 256, 256, 0, stream>>>(spanE,  fB, 256 * 512);
    k_f32_to_bf16<<<(256 * 512 + 255) / 256, 256, 0, stream>>>(imageE, gB, 256 * 512);
    // 2. pack weights into WMMA-B layout (ktShift: 512 rows -> 4, 1024 -> 5)
    k_pack_b<<<(512 * 1024 + 255) / 256, 256, 0, stream>>>(W1,                W1fP, 4);
    k_pack_b<<<(512 * 1024 + 255) / 256, 256, 0, stream>>>(W1 + 512 * 1024,   W1sP, 4);
    k_pack_b<<<(512 * 1024 + 255) / 256, 256, 0, stream>>>(W1 + 1024 * 1024,  W1pP, 4);
    k_pack_b<<<(1024 * 1024 + 255) / 256, 256, 0, stream>>>(W2,               W2P,  5);
    // 3. hf / hs
    k_hfhs<<<256, 256, 0, stream>>>(fB, gB, W1fP, W1sP, hf, hs);
    // 4. fused pairwise MLP -> scores  (32 pairs per block)
    k_main<<<2048, 256, 0, stream>>>(spanE, imageE, hf, hs, b1, b2, W3, b3,
                                     spanM, imageM, W1pP, W2P, scores_out);
    // 5. loss
    k_loss<<<1, 256, 0, stream>>>(scores_out, loss_out);
}
